// FlashAttention_60653528154551
// MI455X (gfx1250) — matrix operand order repaired and checked by reading, not compiled, not run on hardware
//
#include <hip/hip_runtime.h>

typedef __attribute__((ext_vector_type(16))) _Float16 v16h;
typedef __attribute__((ext_vector_type(8)))  _Float16 h8;
typedef __attribute__((ext_vector_type(8)))  float    v8f;
typedef unsigned int u32x4 __attribute__((ext_vector_type(4)));
typedef int          i32x4 __attribute__((ext_vector_type(4)));
typedef int          i32x8 __attribute__((ext_vector_type(8)));

#define DIM    1024
#define HEADS  16
#define DHEAD  64
#define SEQ    2048
#define BATCH  2
#define ROWS   (BATCH*SEQ)   // 4096
#define QKVW   (3*DIM)       // 3072
#define SCALE  0.125f        // 64^-0.5

#if __has_builtin(__builtin_amdgcn_tensor_load_to_lds)
#define HAS_TDM 1
#endif

// 16-byte async global->LDS copy (per lane), tracked by ASYNCcnt.
// Inline asm: the clang builtin's params use the CUDA __device__/__shared__
// language address spaces, which HIP generic pointers can't convert to.
static __device__ __forceinline__ void cp16_async(_Float16* l, const _Float16* g) {
  unsigned loff = (unsigned)(unsigned long long)(void*)l;
  asm volatile("global_load_async_to_lds_b128 %0, %1, off"
               :: "v"(loff), "v"((unsigned long long)(const void*)g)
               : "memory");
}

// TDM: 2D tile load (row-major, 2-byte elements) via Tensor DMA descriptor.
// D# per CDNA5 ISA sec 8.3/8.4: group0 = {count=1, lds_addr, global_addr, type=2},
// group1 = {data_size=1(2B), tensor dims, tile dims, dim0 stride}.
static __device__ __forceinline__ void tdm_load_2d(
    unsigned lds_off, const _Float16* gbase, unsigned tensor_w,
    unsigned tensor_h, unsigned tile_w, unsigned tile_h, unsigned row_stride) {
#if defined(HAS_TDM)
  unsigned long long ga = (unsigned long long)(const void*)gbase;
  u32x4 g0;
  g0[0] = 1u;                                   // count=1 (user descriptor)
  g0[1] = lds_off;                              // lds_addr (bytes)
  g0[2] = (unsigned)(ga & 0xFFFFFFFFu);         // global_addr[31:0]
  g0[3] = (unsigned)((ga >> 32) & 0x01FFFFFFu) | (2u << 30);  // addr hi | type=2
  i32x8 g1;
  g1[0] = (int)(1u << 16);                                    // data_size=1 (2B)
  g1[1] = (int)((tensor_w & 0xFFFFu) << 16);                  // tensor_dim0 lo
  g1[2] = (int)(((tensor_w >> 16) & 0xFFFFu) | ((tensor_h & 0xFFFFu) << 16));
  g1[3] = (int)(((tensor_h >> 16) & 0xFFFFu) | ((tile_w & 0xFFFFu) << 16));
  g1[4] = (int)(tile_h & 0xFFFFu);                            // tile_dim1
  g1[5] = (int)row_stride;                                    // dim0 stride lo
  g1[6] = 0;
  g1[7] = 0;
  i32x4 z4 = {0, 0, 0, 0};
#if __clang_major__ >= 23
  i32x8 z8 = {0, 0, 0, 0, 0, 0, 0, 0};
  __builtin_amdgcn_tensor_load_to_lds(g0, g1, z4, z4, z8, 0);
#else
  __builtin_amdgcn_tensor_load_to_lds(g0, g1, z4, z4, 0);
#endif
#else
  (void)lds_off; (void)gbase; (void)tensor_w; (void)tensor_h;
  (void)tile_w; (void)tile_h; (void)row_stride;
#endif
}

static __device__ __forceinline__ void stage_fence(int wave) {
  asm volatile("s_wait_asynccnt 0" ::: "memory");
#if defined(HAS_TDM)
  if (wave == 0) __builtin_amdgcn_s_wait_tensorcnt(0);
#endif
}

// ---------------------------------------------------------------------------
// WMMA fragment helpers (CDNA5 ISA 7.12.2, wave32)
// ---------------------------------------------------------------------------
static __device__ __forceinline__ v16h load_frag_a(const _Float16* base, int ld) {
  const int lane = threadIdx.x & 31;
  const int m = lane & 15, hf = lane >> 4;
  const _Float16* p = base + m * ld + hf * 8;
  h8 lo = *(const h8*)(p);
  h8 hi = *(const h8*)(p + 16);
  v16h f;
#pragma unroll
  for (int j = 0; j < 8; ++j) { f[j] = lo[j]; f[j + 8] = hi[j]; }
  return f;
}

static __device__ __forceinline__ v16h load_frag_bt(const _Float16* base, int ld) {
  const int lane = threadIdx.x & 31;
  const int n = lane & 15, hf = lane >> 4;
  const _Float16* p = base + n * ld + hf * 8;
  h8 lo = *(const h8*)(p);
  h8 hi = *(const h8*)(p + 16);
  v16h f;
#pragma unroll
  for (int j = 0; j < 8; ++j) { f[j] = lo[j]; f[j + 8] = hi[j]; }
  return f;
}

static __device__ __forceinline__ v8f wmma16(v16h a, v16h b, v8f c) {
  return __builtin_amdgcn_wmma_f32_16x16x32_f16(false, a, false, b, (short)0, c,
                                                false, false);
}

// ---------------------------------------------------------------------------
// Kernel 0a: fp32 -> f16 cast.   Kernel 0b: cast + transpose (dst[c][r]).
// ---------------------------------------------------------------------------
__global__ __launch_bounds__(256) void cast_f16_kernel(
    const float* __restrict__ src, _Float16* __restrict__ dst, int n) {
  int i = blockIdx.x * blockDim.x + threadIdx.x;
  if (i < n) dst[i] = (_Float16)src[i];
}

__global__ __launch_bounds__(256) void transpose_cast_kernel(
    const float* __restrict__ src, _Float16* __restrict__ dst, int R, int C) {
  int i = blockIdx.x * blockDim.x + threadIdx.x;
  if (i < R * C) {
    int r = i / C, c = i % C;
    dst[(size_t)c * R + r] = (_Float16)src[i];
  }
}

// ---------------------------------------------------------------------------
// Kernel 1: QKV GEMM (x16[4096,1024] @ w) + fused RoPE. Weights pre-transposed
// wT[n][k]. Double-buffered LDS; A via async-LDS copies, B via TDM (fallback
// async). q,k stored [b,h,n,d]; v stored TRANSPOSED [b,h,d,n] for attention.
// ---------------------------------------------------------------------------
__global__ __launch_bounds__(256) void qkv_rope_kernel(
    const _Float16* __restrict__ x16, const _Float16* __restrict__ wT,
    _Float16* __restrict__ q16, _Float16* __restrict__ k16,
    _Float16* __restrict__ vT16) {
  __shared__ __attribute__((aligned(16))) _Float16 sA[2][128 * 32];
  __shared__ __attribute__((aligned(16))) _Float16 sBt[2][64 * 32];
  const int m0 = blockIdx.x * 128;
  const int n0 = blockIdx.y * 64;
  const int tid = threadIdx.x;
  const int wave = tid >> 5, lane = tid & 31;
  const int nlane = lane & 15, hf = lane >> 4;
  const int a_mi = tid >> 1, a_kk = (tid & 1) * 16;   // A: 2x16B per thread
  const int b_ni = tid >> 2, b_kk = (tid & 3) * 8;    // B: 1x16B per thread

  auto stage = [&](int k0, int buf) {
    const _Float16* ga = x16 + (size_t)(m0 + a_mi) * DIM + k0 + a_kk;
    cp16_async(&sA[buf][a_mi * 32 + a_kk], ga);
    cp16_async(&sA[buf][a_mi * 32 + a_kk + 8], ga + 8);
#if defined(HAS_TDM)
    if (wave == 0)
      tdm_load_2d((unsigned)(unsigned long long)(void*)&sBt[buf][0],
                  wT + (size_t)n0 * DIM + k0, DIM, QKVW, 32, 64, DIM);
#else
    cp16_async(&sBt[buf][b_ni * 32 + b_kk],
               wT + (size_t)(n0 + b_ni) * DIM + k0 + b_kk);
#endif
  };

  v8f acc[4] = {};
  int buf = 0;
  stage(0, 0);
  stage_fence(wave);
  __syncthreads();
  for (int k0 = 0; k0 < DIM; k0 += 32) {
    const int nxt = buf ^ 1;
    if (k0 + 32 < DIM) stage(k0 + 32, nxt);
    v16h a = load_frag_a(&sA[buf][wave * 16 * 32], 32);
    v16h b[4];
#pragma unroll
    for (int nt = 0; nt < 4; ++nt)
      b[nt] = load_frag_bt(&sBt[buf][nt * 16 * 32], 32);
#pragma unroll
    for (int nt = 0; nt < 4; ++nt) acc[nt] = wmma16(a, b[nt], acc[nt]);
    stage_fence(wave);
    __syncthreads();
    buf = nxt;
  }

  const int region = n0 >> 10;          // 0=q, 1=k, 2=v
  const int h = (n0 & 1023) >> 6;       // head (N tile is head-aligned)

  if (region < 2) {  // RoPE: per-lane 2x2 rotation between acc[t] and acc[t+2]
    const float lg = 0.03125f * __logf(10000.0f);  // (2/64)*ln(10000)
#pragma unroll
    for (int r = 0; r < 8; ++r) {
      const int m = m0 + wave * 16 + r + 8 * hf;
      const float pos = (float)(m & (SEQ - 1));
#pragma unroll
      for (int t = 0; t < 2; ++t) {
        const int d = t * 16 + nlane;               // freq index 0..31
        float s, c;
        __sincosf(pos * __expf(-(float)d * lg), &s, &c);
        float x1 = acc[t][r], x2 = acc[t + 2][r];
        acc[t][r]     = x1 * c - x2 * s;
        acc[t + 2][r] = x2 * c + x1 * s;
      }
    }
    _Float16* dst = (region == 0) ? q16 : k16;
#pragma unroll
    for (int t = 0; t < 4; ++t)
#pragma unroll
      for (int r = 0; r < 8; ++r) {
        const int m = m0 + wave * 16 + r + 8 * hf;
        const int b = m >> 11, pos = m & (SEQ - 1);
        const int d = t * 16 + nlane;
        dst[((size_t)(b * HEADS + h) * SEQ + pos) * DHEAD + d] =
            (_Float16)acc[t][r];
      }
  } else {  // V: store transposed [b,h,d,n]
#pragma unroll
    for (int t = 0; t < 4; ++t)
#pragma unroll
      for (int r = 0; r < 8; ++r) {
        const int m = m0 + wave * 16 + r + 8 * hf;
        const int b = m >> 11, pos = m & (SEQ - 1);
        const int d = t * 16 + nlane;
        vT16[((size_t)(b * HEADS + h) * DHEAD + d) * SEQ + pos] =
            (_Float16)acc[t][r];
      }
  }
}

// ---------------------------------------------------------------------------
// Kernel 2: flash attention. 1 block per (b*h, 64-query tile); 128 threads.
// kv tiles double-buffered, staged with async-LDS copies (K row-major,
// V already transposed in global).
// ---------------------------------------------------------------------------
__global__ __launch_bounds__(128) void attn_kernel(
    const _Float16* __restrict__ q16, const _Float16* __restrict__ k16,
    const _Float16* __restrict__ vT16, _Float16* __restrict__ o16) {
  __shared__ __attribute__((aligned(16))) _Float16 sK[2][64 * 64];   // [key][d]
  __shared__ __attribute__((aligned(16))) _Float16 sVt[2][64 * 64];  // [d][key]
  __shared__ __attribute__((aligned(16))) _Float16 sP[4][16 * 64];
  const int tid = threadIdx.x, wave = tid >> 5, lane = tid & 31;
  const int nlane = lane & 15, hf = lane >> 4;
  const int bh = blockIdx.x >> 5;   // 0..31
  const int qt = blockIdx.x & 31;   // 64-row query tile
  const int q0 = qt * 64 + wave * 16;
  const int st_r = tid >> 1, st_c = (tid & 1) * 32;  // staging row + 32-col chunk

  auto stage = [&](int kt, int buf) {
    const _Float16* gk =
        k16 + ((size_t)bh * SEQ + kt * 64 + st_r) * DHEAD + st_c;
    const _Float16* gv =
        vT16 + ((size_t)bh * DHEAD + st_r) * SEQ + kt * 64 + st_c;
#pragma unroll
    for (int j = 0; j < 32; j += 8) {
      cp16_async(&sK[buf][st_r * 64 + st_c + j], gk + j);
      cp16_async(&sVt[buf][st_r * 64 + st_c + j], gv + j);
    }
  };

  v16h aq[2];
  const _Float16* qbase = q16 + ((size_t)bh * SEQ + q0) * DHEAD;
#pragma unroll
  for (int ks = 0; ks < 2; ++ks) aq[ks] = load_frag_a(qbase + ks * 32, DHEAD);

  float mrow[8], lrow[8];
#pragma unroll
  for (int r = 0; r < 8; ++r) { mrow[r] = -3.0e38f; lrow[r] = 0.f; }
  v8f accO[4] = {};

  int buf = 0;
  stage(0, 0);
  asm volatile("s_wait_asynccnt 0" ::: "memory");
  __syncthreads();

  for (int kt = 0; kt < SEQ / 64; ++kt) {
    const int nxt = buf ^ 1;
    if (kt + 1 < SEQ / 64) stage(kt + 1, nxt);

    // S = Q @ K^T (contract over d)
    v8f accS[4] = {};
#pragma unroll
    for (int ks = 0; ks < 2; ++ks) {
      v16h b[4];
#pragma unroll
      for (int nt = 0; nt < 4; ++nt)
        b[nt] = load_frag_bt(&sK[buf][nt * 16 * 64 + ks * 32], 64);
#pragma unroll
      for (int nt = 0; nt < 4; ++nt) accS[nt] = wmma16(aq[ks], b[nt], accS[nt]);
    }

    // online softmax: butterfly over 16-lane halves (rows live per half)
#pragma unroll
    for (int r = 0; r < 8; ++r) {
      float mx = -3.0e38f;
#pragma unroll
      for (int nt = 0; nt < 4; ++nt) {
        float s = accS[nt][r] * SCALE;
        accS[nt][r] = s;
        mx = fmaxf(mx, s);
      }
#pragma unroll
      for (int o = 1; o < 16; o <<= 1) mx = fmaxf(mx, __shfl_xor(mx, o, 32));
      const float newm = fmaxf(mrow[r], mx);
      const float fac = __expf(mrow[r] - newm);
      float rs = 0.f;
#pragma unroll
      for (int nt = 0; nt < 4; ++nt) {
        float p = __expf(accS[nt][r] - newm);
        accS[nt][r] = p;
        rs += p;
      }
#pragma unroll
      for (int o = 1; o < 16; o <<= 1) rs += __shfl_xor(rs, o, 32);
      lrow[r] = lrow[r] * fac + rs;
      mrow[r] = newm;
#pragma unroll
      for (int dt = 0; dt < 4; ++dt) accO[dt][r] *= fac;
    }

    // C-layout -> A-layout transpose of P via per-wave LDS tile
#pragma unroll
    for (int nt = 0; nt < 4; ++nt)
#pragma unroll
      for (int r = 0; r < 8; ++r)
        sP[wave][(r + 8 * hf) * 64 + nt * 16 + nlane] = (_Float16)accS[nt][r];
    asm volatile("s_wait_dscnt 0" ::: "memory");  // wave-local LDS ordering

    // O += P @ V (contract over keys)
#pragma unroll
    for (int ks = 0; ks < 2; ++ks) {
      v16h ap = load_frag_a(&sP[wave][ks * 32], 64);
      v16h b[4];
#pragma unroll
      for (int dt = 0; dt < 4; ++dt)
        b[dt] = load_frag_bt(&sVt[buf][dt * 16 * 64 + ks * 32], 64);
#pragma unroll
      for (int dt = 0; dt < 4; ++dt) accO[dt] = wmma16(ap, b[dt], accO[dt]);
    }

    asm volatile("s_wait_asynccnt 0" ::: "memory");
    __syncthreads();
    buf = nxt;
  }

  // normalize + store to [b*n, h*d] f16 for the projection GEMM
  const int b = bh >> 4, h = bh & 15;
#pragma unroll
  for (int dt = 0; dt < 4; ++dt)
#pragma unroll
    for (int r = 0; r < 8; ++r) {
      const float v = accO[dt][r] / lrow[r];
      const int row = b * SEQ + q0 + r + 8 * hf;
      const int col = h * DHEAD + dt * 16 + nlane;
      o16[(size_t)row * (HEADS * DHEAD) + col] = (_Float16)v;
    }
}

// ---------------------------------------------------------------------------
// Kernel 3: out projection GEMM (+bias, fp32 out). Weights pre-transposed.
// ---------------------------------------------------------------------------
__global__ __launch_bounds__(256) void proj_kernel(
    const _Float16* __restrict__ a16, const _Float16* __restrict__ wT,
    const float* __restrict__ bias, float* __restrict__ out) {
  __shared__ __attribute__((aligned(16))) _Float16 sA[2][128 * 32];
  __shared__ __attribute__((aligned(16))) _Float16 sBt[2][64 * 32];
  const int m0 = blockIdx.x * 128;
  const int n0 = blockIdx.y * 64;
  const int tid = threadIdx.x;
  const int wave = tid >> 5, lane = tid & 31;
  const int nlane = lane & 15, hf = lane >> 4;
  const int a_mi = tid >> 1, a_kk = (tid & 1) * 16;
  const int b_ni = tid >> 2, b_kk = (tid & 3) * 8;

  auto stage = [&](int k0, int buf) {
    const _Float16* ga = a16 + (size_t)(m0 + a_mi) * DIM + k0 + a_kk;
    cp16_async(&sA[buf][a_mi * 32 + a_kk], ga);
    cp16_async(&sA[buf][a_mi * 32 + a_kk + 8], ga + 8);
#if defined(HAS_TDM)
    if (wave == 0)
      tdm_load_2d((unsigned)(unsigned long long)(void*)&sBt[buf][0],
                  wT + (size_t)n0 * DIM + k0, DIM, DIM, 32, 64, DIM);
#else
    cp16_async(&sBt[buf][b_ni * 32 + b_kk],
               wT + (size_t)(n0 + b_ni) * DIM + k0 + b_kk);
#endif
  };

  v8f acc[4] = {};
  int buf = 0;
  stage(0, 0);
  stage_fence(wave);
  __syncthreads();
  for (int k0 = 0; k0 < DIM; k0 += 32) {
    const int nxt = buf ^ 1;
    if (k0 + 32 < DIM) stage(k0 + 32, nxt);
    v16h a = load_frag_a(&sA[buf][wave * 16 * 32], 32);
    v16h b[4];
#pragma unroll
    for (int nt = 0; nt < 4; ++nt)
      b[nt] = load_frag_bt(&sBt[buf][nt * 16 * 32], 32);
#pragma unroll
    for (int nt = 0; nt < 4; ++nt) acc[nt] = wmma16(a, b[nt], acc[nt]);
    stage_fence(wave);
    __syncthreads();
    buf = nxt;
  }
#pragma unroll
  for (int nt = 0; nt < 4; ++nt)
#pragma unroll
    for (int r = 0; r < 8; ++r) {
      const int m = m0 + wave * 16 + r + 8 * hf;
      const int n = n0 + nt * 16 + nlane;
      out[(size_t)m * DIM + n] = acc[nt][r] + bias[n];
    }
}

// ---------------------------------------------------------------------------
// Launch. Workspace (48 MiB):
//   [ 0, 8) x16   [ 8,14) wqkvT16   [14,16) woutT16
//   [16,24) q16   [24,32) k16       [32,40) vT16      [40,48) attn16
// ---------------------------------------------------------------------------
extern "C" void kernel_launch(void* const* d_in, const int* in_sizes, int n_in,
                              void* d_out, int out_size, void* d_ws,
                              size_t ws_size, hipStream_t stream) {
  (void)in_sizes; (void)n_in; (void)out_size; (void)ws_size;
  const float* x     = (const float*)d_in[0];
  const float* qkv_w = (const float*)d_in[1];
  const float* out_w = (const float*)d_in[2];
  const float* out_b = (const float*)d_in[3];
  float* out = (float*)d_out;

  char* ws = (char*)d_ws;
  const size_t MB = 1ull << 20;
  _Float16* x16     = (_Float16*)(ws + 0 * MB);
  _Float16* wqkvT16 = (_Float16*)(ws + 8 * MB);
  _Float16* woutT16 = (_Float16*)(ws + 14 * MB);
  _Float16* q16     = (_Float16*)(ws + 16 * MB);
  _Float16* k16     = (_Float16*)(ws + 24 * MB);
  _Float16* vT16    = (_Float16*)(ws + 32 * MB);
  _Float16* ao16    = (_Float16*)(ws + 40 * MB);

  const int nx = ROWS * DIM, nwq = DIM * QKVW, nwo = DIM * DIM;
  cast_f16_kernel<<<nx / 256, 256, 0, stream>>>(x, x16, nx);
  transpose_cast_kernel<<<nwq / 256, 256, 0, stream>>>(qkv_w, wqkvT16, DIM, QKVW);
  transpose_cast_kernel<<<nwo / 256, 256, 0, stream>>>(out_w, woutT16, DIM, DIM);

  qkv_rope_kernel<<<dim3(ROWS / 128, QKVW / 64), 256, 0, stream>>>(
      x16, wqkvT16, q16, k16, vT16);

  attn_kernel<<<BATCH * HEADS * (SEQ / 64), 128, 0, stream>>>(q16, k16, vT16,
                                                              ao16);

  proj_kernel<<<dim3(ROWS / 128, DIM / 64), 256, 0, stream>>>(ao16, woutT16,
                                                              out_b, out);
}
